// HyperbolicAttentionHardened_25615184953643
// MI455X (gfx1250) — matrix operand order repaired and checked by reading, not compiled, not run on hardware
//
#include <hip/hip_runtime.h>

// ---------------------------------------------------------------------------
// HyperbolicAttention for MI455X (gfx1250, wave32, WMMA 16x16x32 f16->f32)
// B=2 L=2048 E=1024 H=16 HD=64.  Flash attention, analytic causal mask
// (never reads the 16MB attention_mask input), f16 operands / f32 accum.
// GEMM tiles staged with CDNA5 async global->LDS copies (ASYNCcnt).
// ---------------------------------------------------------------------------

typedef __attribute__((ext_vector_type(16))) _Float16 v16h;
typedef __attribute__((ext_vector_type(8)))  float    v8f;

#define DEVINL static __device__ __forceinline__

static constexpr int BB = 2;
static constexpr int LL = 2048;
static constexpr int EE = 1024;
static constexpr int HH = 16;
static constexpr int HD = 64;
static constexpr int MM = BB * LL;          // 4096 rows
static constexpr int LDO = EE + 1;          // 1025 output row stride

DEVINL float sanitize(float s) {
  // torch.nan_to_num(x, nan=0, posinf=1e4, neginf=-1e4)
  if (__builtin_isnan(s)) return 0.f;
  if (__builtin_isinf(s)) return s > 0.f ? 1e4f : -1e4f;
  return s;
}

DEVINL v8f wmma16(v16h a, v16h b, v8f c) {
  // v_wmma_f32_16x16x32_f16: (neg_a, A, neg_b, B, c_mod, C, reuse_a, reuse_b)
  return __builtin_amdgcn_wmma_f32_16x16x32_f16(false, a, false, b, (short)0, c,
                                                false, false);
}

// CDNA5 async global->LDS copy, 16 bytes per lane, tracked by ASYNCcnt.
// lds_off = LDS byte address (low 32 bits of the flat shared address,
// per ISA 10.2 aperture rules).
DEVINL void async_copy_b128(unsigned lds_off, const _Float16* gaddr) {
  asm volatile("global_load_async_to_lds_b128 %0, %1, off"
               :: "v"(lds_off), "v"(gaddr)
               : "memory");
}

// A fragment (M=16 x K=32, f16). Contiguous k order (not ISA table order): lane holds row m=lane&15;
// lanes 0-15 cover K 0..15; lanes 16-31 cover K 16..31;
// consecutive K pairs packed per 32-bit VGPR.
DEVINL v16h load_afrag(const _Float16* base, int ld, int lane, int koff) {
  const _Float16* row =
      base + (size_t)(lane & 15) * ld + koff + ((lane >> 4) << 4);
  union { v16h h; unsigned u[8]; } f;
#pragma unroll
  for (int j = 0; j < 4; ++j) {
    f.u[j]     = *(const unsigned*)(row + 2 * j);        // K = kg+2j, kg+2j+1
    f.u[4 + j] = *(const unsigned*)(row + 8 + 2 * j);    // K = 8+kg+2j, +1
  }
  return f.h;
}

// B fragment (K=32 x N=16, f16) where the SOURCE is row-major [N][K]
// (weight rows W[n][k], or K-matrix rows).  Same contiguous k order as A: lane = N column
// (lane&15); lanes 0-15 hold K=0..15, lanes 16-31 hold K=16..31,
// consecutive K pairs per VGPR.
DEVINL v16h load_bfrag(const _Float16* base, int ld, int lane, int koff) {
  const _Float16* row =
      base + (size_t)(lane & 15) * ld + koff + ((lane >> 4) << 4);
  union { v16h h; unsigned u[8]; } f;
#pragma unroll
  for (int r = 0; r < 8; ++r)
    f.u[r] = *(const unsigned*)(row + 2 * r);
  return f.h;
}

// B fragment for P@V: B[k=key][n=d] with V stored row-major [key][d];
// consecutive-K pairs live in different V rows, assemble from two u16 loads.
DEVINL v16h load_vfrag(const _Float16* Vp, int ldv, int key0, int dbase,
                       int lane) {
  int d  = dbase + (lane & 15);
  int kg = key0 + ((lane >> 4) << 4);
  union { v16h h; unsigned u[8]; } f;
#pragma unroll
  for (int r = 0; r < 8; ++r) {
    unsigned lo = *(const unsigned short*)(Vp + (size_t)(kg + 2 * r) * ldv + d);
    unsigned hi = *(const unsigned short*)(Vp + (size_t)(kg + 2 * r + 1) * ldv + d);
    f.u[r] = lo | (hi << 16);
  }
  return f.h;
}

// reductions across the 16 lanes holding one matrix row (wave32 halves)
DEVINL float rmax16(float v) {
  v = fmaxf(v, __shfl_xor(v, 1, 32));
  v = fmaxf(v, __shfl_xor(v, 2, 32));
  v = fmaxf(v, __shfl_xor(v, 4, 32));
  v = fmaxf(v, __shfl_xor(v, 8, 32));
  return v;
}
DEVINL float rsum16(float v) {
  v += __shfl_xor(v, 1, 32);
  v += __shfl_xor(v, 2, 32);
  v += __shfl_xor(v, 4, 32);
  v += __shfl_xor(v, 8, 32);
  return v;
}

// ---------------------------------------------------------------------------
// conversion kernels
// ---------------------------------------------------------------------------
__global__ __launch_bounds__(256) void cvt_x_kernel(const float* __restrict__ x,
                                                    _Float16* __restrict__ Xh) {
  int i = blockIdx.x * 256 + threadIdx.x;          // over MM*EE
  if (i >= MM * EE) return;
  int row = i >> 10, col = i & (EE - 1);
  float s = sanitize(x[(size_t)row * LDO + 1 + col]);   // drop time column
  Xh[i] = (_Float16)s;
}

__global__ __launch_bounds__(256) void cvt_w_kernel(const float* __restrict__ w,
                                                    _Float16* __restrict__ wh,
                                                    int n) {
  int i = blockIdx.x * 256 + threadIdx.x;
  if (i < n) wh[i] = (_Float16)w[i];
}

__global__ __launch_bounds__(256) void zero_kernel(float* __restrict__ p, int n) {
  int i = blockIdx.x * 256 + threadIdx.x;
  if (i < n) p[i] = 0.f;
}

__global__ __launch_bounds__(256) void time_kernel(const float* __restrict__ normsq,
                                                   const float* __restrict__ Kc,
                                                   float* __restrict__ out) {
  int i = blockIdx.x * 256 + threadIdx.x;
  if (i < MM) out[(size_t)i * LDO] = sqrtf(1.f / Kc[0] + normsq[i] + 1e-8f);
}

// ---------------------------------------------------------------------------
// GEMM: C[M,N] = A[M,K] * W^T  (A row-major [M][K] f16, W row-major [N][K] f16)
// Block tile 128x128, 8 waves (4M x 2N), per-wave 32x64 (2x4 WMMA frags),
// K step 32.  Tiles staged via global_load_async_to_lds_b128 (ASYNCcnt) with
// global_prefetch of the next K-step.  MODE 0: f16 out [M][1024].
// MODE 1: f32 out into d_out (stride 1025, +1 col) + row-norm atomics.
// ---------------------------------------------------------------------------
template <int MODE>
__global__ __launch_bounds__(256) void gemm_kernel(const _Float16* __restrict__ A,
                                                   const _Float16* __restrict__ Bw,
                                                   void* __restrict__ Cout,
                                                   float* __restrict__ normsq) {
  __shared__ _Float16 As[128][48];   // stride 48 f16 = 96B (16B aligned rows)
  __shared__ _Float16 Bs[128][48];

  const int tid = threadIdx.x, lane = tid & 31, wid = tid >> 5;
  const int wm = wid & 3, wn = wid >> 2;
  const int Mbase = blockIdx.y * 128, Nbase = blockIdx.x * 128;
  const int lr = tid >> 1;               // tile row 0..127 for staging
  const int lk = (tid & 1) * 16;         // k half (16 f16 = 32B)

  // LDS byte offsets for this thread's async-copy destinations
  const unsigned la = (unsigned)(size_t)&As[lr][lk];
  const unsigned lb = (unsigned)(size_t)&Bs[lr][lk];
  const _Float16* agr = A  + (size_t)(Mbase + lr) * EE + lk;
  const _Float16* bgr = Bw + (size_t)(Nbase + lr) * EE + lk;

  v8f acc[2][4] = {};

  for (int kt = 0; kt < EE; kt += 32) {
    __syncthreads();                       // previous iter's LDS reads done
    const _Float16* ag = agr + kt;
    const _Float16* bg = bgr + kt;
    async_copy_b128(la,      ag);          // A tile: 2 x 16B per thread
    async_copy_b128(la + 16, ag + 8);
    async_copy_b128(lb,      bg);          // B tile: 2 x 16B per thread
    async_copy_b128(lb + 16, bg + 8);
    if (kt + 32 < EE) {                    // warm L2/WGP$ for next K-step
      __builtin_prefetch(ag + 32, 0, 0);
      __builtin_prefetch(bg + 32, 0, 0);
    }
    asm volatile("s_wait_asynccnt 0" ::: "memory");
    __syncthreads();                       // tiles visible to all waves

    v16h af0 = load_afrag(&As[wm * 32][0],      48, lane, 0);
    v16h af1 = load_afrag(&As[wm * 32 + 16][0], 48, lane, 0);
    v16h bf[4];
#pragma unroll
    for (int j = 0; j < 4; ++j)
      bf[j] = load_bfrag(&Bs[wn * 64 + j * 16][0], 48, lane, 0);
#pragma unroll
    for (int j = 0; j < 4; ++j) {
      acc[0][j] = wmma16(af0, bf[j], acc[0][j]);
      acc[1][j] = wmma16(af1, bf[j], acc[1][j]);
    }
  }

  // epilogue: C/D layout = VGPR e: row e + 8*(lane>>4), col lane&15
  const int half = lane >> 4, nl = lane & 15;
  if (MODE == 0) {
    _Float16* C = (_Float16*)Cout;
#pragma unroll
    for (int i = 0; i < 2; ++i)
#pragma unroll
      for (int e = 0; e < 8; ++e) {
        int row = Mbase + wm * 32 + i * 16 + e + half * 8;
#pragma unroll
        for (int j = 0; j < 4; ++j) {
          int col = Nbase + wn * 64 + j * 16 + nl;
          C[(size_t)row * EE + col] = (_Float16)acc[i][j][e];
        }
      }
  } else {
    float* C = (float*)Cout;
#pragma unroll
    for (int i = 0; i < 2; ++i)
#pragma unroll
      for (int e = 0; e < 8; ++e) {
        int row = Mbase + wm * 32 + i * 16 + e + half * 8;
        float ss = 0.f;
#pragma unroll
        for (int j = 0; j < 4; ++j) {
          int col = Nbase + wn * 64 + j * 16 + nl;
          float v = acc[i][j][e];
          C[(size_t)row * LDO + 1 + col] = v;
          ss += v * v;
        }
        ss = rsum16(ss);
        if (nl == 0) atomicAdd(&normsq[row], ss);
      }
  }
}

// ---------------------------------------------------------------------------
// Flash attention: one wave per 16-query block per (b,h). Key blocks of 32.
// scores: 4x v_wmma (2 key tiles x K-dim 64 chained); PV: 4x v_wmma.
// P reshuffled C-layout -> A-layout through per-wave LDS (s_wait_dscnt 0).
// Masked entries set to -50 (matches reference's clip(mask+scores)).
// ---------------------------------------------------------------------------
__global__ __launch_bounds__(256) void attn_kernel(const _Float16* __restrict__ Qh,
                                                   const _Float16* __restrict__ Kh,
                                                   const _Float16* __restrict__ Vh,
                                                   _Float16* __restrict__ Oh) {
  __shared__ _Float16 Pbuf[8][16][34];    // per-wave 16x32 (+pad) P staging

  const int tid = threadIdx.x, lane = tid & 31, wid = tid >> 5;
  const int half = lane >> 4, nl = lane & 15;
  const int gw  = blockIdx.x * 8 + wid;   // 4096 waves total
  const int b   = gw >> 11;               // / (H * L/16)
  const int rem = gw & 2047;
  const int h   = rem >> 7;
  const int q0  = (rem & 127) * 16;       // query block start (local l)

  const size_t rowbase = (size_t)b * LL + q0;
  const _Float16* Qp    = Qh + rowbase * EE + h * HD;
  const _Float16* Kbase = Kh + ((size_t)b * LL) * EE + h * HD;
  const _Float16* Vbase = Vh + ((size_t)b * LL) * EE + h * HD;

  // Q fragments held for the whole wave (A-layout, direct from global)
  const v16h qa0 = load_afrag(Qp, EE, lane, 0);
  const v16h qa1 = load_afrag(Qp, EE, lane, 32);

  v8f o[4] = {};
  float mrow[8], lrow[8];
#pragma unroll
  for (int e = 0; e < 8; ++e) { mrow[e] = -1e30f; lrow[e] = 0.f; }

  _Float16* pb = &Pbuf[wid][0][0];
  const int nkb = (q0 >> 5) + 1;          // causal: key blocks of 32
  for (int kb = 0; kb < nkb; ++kb) {
    const int key0 = kb * 32;

    // ---- scores: S0 (keys key0..+15), S1 (keys key0+16..+31) ----
    v8f s0 = {}, s1 = {};
    {
      v16h k0a = load_bfrag(Kbase + (size_t)key0 * EE, EE, lane, 0);
      v16h k0b = load_bfrag(Kbase + (size_t)key0 * EE, EE, lane, 32);
      s0 = wmma16(qa0, k0a, s0);
      s0 = wmma16(qa1, k0b, s0);
      v16h k1a = load_bfrag(Kbase + (size_t)(key0 + 16) * EE, EE, lane, 0);
      v16h k1b = load_bfrag(Kbase + (size_t)(key0 + 16) * EE, EE, lane, 32);
      s1 = wmma16(qa0, k1a, s1);
      s1 = wmma16(qa1, k1b, s1);
    }

    // ---- scale, causal mask, clip, online softmax ----
    float alpha[8];
#pragma unroll
    for (int e = 0; e < 8; ++e) {
      const int qi = q0 + e + half * 8;          // local query index
      float a = (float)s0[e] * 0.125f;           // 1/sqrt(64)
      a = (key0 + nl > qi) ? -50.f : fminf(fmaxf(a, -50.f), 50.f);
      float c = (float)s1[e] * 0.125f;
      c = (key0 + 16 + nl > qi) ? -50.f : fminf(fmaxf(c, -50.f), 50.f);

      const float mnew = fmaxf(mrow[e], rmax16(fmaxf(a, c)));
      alpha[e] = __expf(mrow[e] - mnew);
      const float p0 = __expf(a - mnew);
      const float p1 = __expf(c - mnew);
      lrow[e] = lrow[e] * alpha[e] + rsum16(p0 + p1);
      mrow[e] = mnew;

      const int prow = e + half * 8;             // C-layout row in P tile
      pb[prow * 34 + nl]      = (_Float16)p0;
      pb[prow * 34 + 16 + nl] = (_Float16)p1;
    }

    // rescale running output
#pragma unroll
    for (int nt = 0; nt < 4; ++nt)
#pragma unroll
      for (int e = 0; e < 8; ++e) o[nt][e] *= alpha[e];

    // wave-local LDS fence before re-reading P in A-layout
    asm volatile("s_wait_dscnt 0" ::: "memory");
    const v16h pa = load_afrag(pb, 34, lane, 0);

    // ---- O += P @ V ----
#pragma unroll
    for (int nt = 0; nt < 4; ++nt) {
      v16h vb = load_vfrag(Vbase, EE, key0, nt * 16, lane);
      o[nt] = wmma16(pa, vb, o[nt]);
    }
  }

  // ---- normalize + store head output (f16, [4096][1024] layout) ----
  _Float16* Orow = Oh + rowbase * EE + h * HD;
#pragma unroll
  for (int e = 0; e < 8; ++e) {
    const float inv = 1.f / lrow[e];
    const int row = e + half * 8;
#pragma unroll
    for (int nt = 0; nt < 4; ++nt)
      Orow[(size_t)row * EE + nt * 16 + nl] = (_Float16)(o[nt][e] * inv);
  }
}

// ---------------------------------------------------------------------------
// launch
// ---------------------------------------------------------------------------
extern "C" void kernel_launch(void* const* d_in, const int* in_sizes, int n_in,
                              void* d_out, int out_size, void* d_ws, size_t ws_size,
                              hipStream_t stream) {
  (void)in_sizes; (void)n_in; (void)out_size; (void)ws_size;
  const float* x  = (const float*)d_in[0];
  const float* Wq = (const float*)d_in[1];
  const float* Wk = (const float*)d_in[2];
  const float* Wv = (const float*)d_in[3];
  const float* Wo = (const float*)d_in[4];
  // d_in[5] = attention_mask: unused (causal mask computed analytically)
  const float* Kc = (const float*)d_in[6];
  float* out = (float*)d_out;

  // workspace layout (~48 MB, 256B aligned slices)
  char* ws = (char*)d_ws;
  size_t off = 0;
  auto carve = [&](size_t bytes) -> char* {
    char* p = ws + off;
    off += (bytes + 255) & ~(size_t)255;
    return p;
  };
  const size_t actb = (size_t)MM * EE * sizeof(_Float16);    // 8 MB
  const size_t wb   = (size_t)EE * EE * sizeof(_Float16);    // 2 MB
  _Float16* Xh  = (_Float16*)carve(actb);
  _Float16* Wqh = (_Float16*)carve(wb);
  _Float16* Wkh = (_Float16*)carve(wb);
  _Float16* Wvh = (_Float16*)carve(wb);
  _Float16* Woh = (_Float16*)carve(wb);
  _Float16* Qh  = (_Float16*)carve(actb);
  _Float16* Kh  = (_Float16*)carve(actb);
  _Float16* Vh  = (_Float16*)carve(actb);
  _Float16* Oh  = (_Float16*)carve(actb);
  float*    normsq = (float*)carve((size_t)MM * sizeof(float));

  // 1) sanitize + downconvert
  cvt_x_kernel<<<(MM * EE) / 256, 256, 0, stream>>>(x, Xh);
  cvt_w_kernel<<<(EE * EE) / 256, 256, 0, stream>>>(Wq, Wqh, EE * EE);
  cvt_w_kernel<<<(EE * EE) / 256, 256, 0, stream>>>(Wk, Wkh, EE * EE);
  cvt_w_kernel<<<(EE * EE) / 256, 256, 0, stream>>>(Wv, Wvh, EE * EE);
  cvt_w_kernel<<<(EE * EE) / 256, 256, 0, stream>>>(Wo, Woh, EE * EE);

  // 2) Q/K/V projections (xs @ W^T), f16 out
  dim3 ggrid(EE / 128, MM / 128);   // (8, 32)
  gemm_kernel<0><<<ggrid, 256, 0, stream>>>(Xh, Wqh, Qh, nullptr);
  gemm_kernel<0><<<ggrid, 256, 0, stream>>>(Xh, Wkh, Kh, nullptr);
  gemm_kernel<0><<<ggrid, 256, 0, stream>>>(Xh, Wvh, Vh, nullptr);

  // 3) causal flash attention (4096 waves, 8 per block)
  attn_kernel<<<MM * HH / (16 * 8), 256, 0, stream>>>(Qh, Kh, Vh, Oh);

  // 4) output projection -> d_out cols 1..1024 + row-norm accumulation
  zero_kernel<<<MM / 256, 256, 0, stream>>>(normsq, MM);
  gemm_kernel<1><<<ggrid, 256, 0, stream>>>(Oh, Woh, out, normsq);

  // 5) time component -> d_out col 0
  time_kernel<<<MM / 256, 256, 0, stream>>>(normsq, Kc, out);
}
